// GraphEncoder_24120536334825
// MI455X (gfx1250) — matrix operand order repaired and checked by reading, not compiled, not run on hardware
//
#include <hip/hip_runtime.h>
#include <hip/hip_bf16.h>
#include <math.h>

#define N_NODES 50000
#define N_EDGES 800000
#define E_TOT   (N_EDGES + N_NODES)   // self-loops appended, every segment non-empty
#define EMB     384
#define HID     64
#define HEADS   4
#define C1      (HEADS * HID)          // 256
#define KP_PAD  8                      // LDS panel pad (halves) -> 16B, breaks bank conflicts
#define MAX_KP  (EMB + KP_PAD)         // 392

typedef __attribute__((ext_vector_type(16))) _Float16 v16h;
typedef __attribute__((ext_vector_type(8)))  float    v8f;

union FragU { uint4 u[2]; v16h v; };

// A fragment: 16x32 f16, row-major A, lane = halfid*16 + (row%16)
// elements 0..7 = K[k0 + half*8 .. +7], elements 8..15 = K[k0 + half*8 + 16 .. +23]
__device__ __forceinline__ v16h load_a_frag(const _Float16* rowk0, int halfid) {
  const _Float16* p = rowk0 + halfid * 8;
  FragU t;
  t.u[0] = *reinterpret_cast<const uint4*>(p);
  t.u[1] = *reinterpret_cast<const uint4*>(p + 16);
  return t.v;
}
// B fragment: 32x16 f16, panel stored [col][Kp]; lane holds col n = lane%16,
// elements 0..7 = K[k0 + half*8 .. +7], elements 8..15 = K[k0 + half*8 + 16 .. +23]
__device__ __forceinline__ v16h load_b_frag(const _Float16* colk0, int halfid) {
  const _Float16* p = colk0 + halfid * 8;
  FragU t;
  t.u[0] = *reinterpret_cast<const uint4*>(p);
  t.u[1] = *reinterpret_cast<const uint4*>(p + 16);
  return t.v;
}

// C[M,N] = A[M,K] @ B[K,N]; A f16 row-major, Bt f16 [N][K]; M%16==0, N%64==0, K%32==0.
// Block = 256 threads (8 waves); gridDim.y = N/64 selects the 64-column panel,
// gridDim.x covers M tiles (one 16x64 strip per wave).
// B panel staged into LDS with CDNA5 async global->LDS copies (ASYNCcnt).
__global__ void ge_gemm_wmma(const _Float16* __restrict__ A,
                             const _Float16* __restrict__ Bt,
                             float* __restrict__ C,
                             const float* __restrict__ bias,  // per-column, may be null
                             int M, int N, int K, int relu) {
  __shared__ __align__(16) _Float16 sB[64 * MAX_KP];

  const int lane   = threadIdx.x & 31;
  const int wave   = threadIdx.x >> 5;
  const int halfid = lane >> 4;
  const int l16    = lane & 15;
  const int Kp     = K + KP_PAD;

  // ---- stage B panel: 64 columns x K halves, padded stride Kp ----
  {
    const int n0 = blockIdx.y * 64;
    const int chunksPerCol = K >> 3;               // 16B chunks per column
    const int totChunks    = chunksPerCol << 6;    // * 64 columns
    for (int ci = threadIdx.x; ci < totChunks; ci += blockDim.x) {
      const int col = ci / chunksPerCol;
      const int within = ci - col * chunksPerCol;
      const unsigned long long gaddr =
          (unsigned long long)(const void*)(Bt + (size_t)(n0 + col) * K + within * 8);
      // Low 32 bits of the flat address of an LDS object = LDS byte offset
      // (ISA 10.2); deriving it from &sB also escapes sB so the "memory"
      // clobber makes the asm a visible writer of sB.
      const unsigned ldsOff =
          (unsigned)(size_t)(&sB[col * Kp + within * 8]);
      // CDNA5 async copy global -> LDS (tracked with ASYNCcnt)
      asm volatile("global_load_async_to_lds_b128 %0, %1, off"
                   :: "v"(ldsOff), "v"(gaddr) : "memory");
    }
    asm volatile("s_wait_asynccnt 0x0" ::: "memory");
  }
  __syncthreads();

  const int mtiles = M >> 4;
  const int tile   = blockIdx.x * 8 + wave;
  if (tile >= mtiles) return;                // wave-uniform, after the barrier
  const int mt  = tile;
  const int row = mt * 16 + l16;             // A row read by this lane
  const int n0  = blockIdx.y * 64;

  v8f acc[4] = {};
  const _Float16* arow = A + (size_t)row * K;
  const _Float16* b0 = sB + (size_t)(0 * 16 + l16) * Kp;
  const _Float16* b1 = sB + (size_t)(1 * 16 + l16) * Kp;
  const _Float16* b2 = sB + (size_t)(2 * 16 + l16) * Kp;
  const _Float16* b3 = sB + (size_t)(3 * 16 + l16) * Kp;

  auto compute_k = [&](v16h afv, int kk) {
    // issue all 8 ds_load_b128 before any WMMA -> single dscnt wait per k-step
    v16h bf0 = load_b_frag(b0 + kk, halfid);
    v16h bf1 = load_b_frag(b1 + kk, halfid);
    v16h bf2 = load_b_frag(b2 + kk, halfid);
    v16h bf3 = load_b_frag(b3 + kk, halfid);
    acc[0] = __builtin_amdgcn_wmma_f32_16x16x32_f16(
        false, afv, false, bf0, (short)0, acc[0], false, false);
    acc[1] = __builtin_amdgcn_wmma_f32_16x16x32_f16(
        false, afv, false, bf1, (short)0, acc[1], false, false);
    acc[2] = __builtin_amdgcn_wmma_f32_16x16x32_f16(
        false, afv, false, bf2, (short)0, acc[2], false, false);
    acc[3] = __builtin_amdgcn_wmma_f32_16x16x32_f16(
        false, afv, false, bf3, (short)0, acc[3], false, false);
  };

  // software-pipeline the A fragment (global); B fragments come from LDS
  v16h af = load_a_frag(arow, halfid);
  int k0 = 0;
  for (; k0 + 32 < K; k0 += 32) {
    v16h afn = load_a_frag(arow + k0 + 32, halfid);   // prefetch next A
    compute_k(af, k0);
    af = afn;
  }
  compute_k(af, k0);                                   // peeled last step

#pragma unroll
  for (int t = 0; t < 4; ++t) {
    const int col = n0 + t * 16 + l16;
#pragma unroll
    for (int v = 0; v < 8; ++v) {
      const int r = mt * 16 + v + halfid * 8;
      float val = acc[t][v];
      if (bias) val += bias[col];
      if (relu) val = fmaxf(val, 0.0f);
      C[(size_t)r * N + col] = val;
    }
  }
}

__global__ void ge_fill(float* p, float v, long long n) {
  long long i = (long long)blockIdx.x * blockDim.x + threadIdx.x;
  if (i < n) p[i] = v;
}

__global__ void ge_cvt_f16(const float* __restrict__ in, _Float16* __restrict__ out, long long n) {
  long long i = (long long)blockIdx.x * blockDim.x + threadIdx.x;
  if (i < n) out[i] = (_Float16)in[i];
}

// W[K,N] f32 -> Wt[N,K] f16
__global__ void ge_pack_wt(const float* __restrict__ W, _Float16* __restrict__ Wt, int K, int N) {
  long long i = (long long)blockIdx.x * blockDim.x + threadIdx.x;
  if (i >= (long long)K * N) return;
  int k = (int)(i / N), n = (int)(i % N);
  Wt[(size_t)n * K + k] = (_Float16)W[i];
}

// per-node, per-head attention logits: als[n,h] = <H[n,h,:], as[h,:]>
__global__ void ge_logits(const float* __restrict__ H, const float* __restrict__ as_,
                          const float* __restrict__ ad_, float* __restrict__ als,
                          float* __restrict__ ald, int Nn, int heads) {
  int i = blockIdx.x * blockDim.x + threadIdx.x;
  if (i >= Nn * heads) return;
  int n = i / heads, h = i % heads;
  const float* hp = H + ((size_t)n * heads + h) * HID;
  const float* sp = as_ + h * HID;
  const float* dp = ad_ + h * HID;
  float s = 0.f, d = 0.f;
  for (int c = 0; c < HID; ++c) { s += hp[c] * sp[c]; d += hp[c] * dp[c]; }
  als[i] = s; ald[i] = d;
}

__device__ __forceinline__ void ge_edge(const int* ei, int e, int& s, int& d) {
  if (e < N_EDGES) { s = ei[e]; d = ei[N_EDGES + e]; }
  else             { s = d = e - N_EDGES; }
}
__device__ __forceinline__ float ge_leaky(float x) { return x > 0.f ? x : 0.2f * x; }

__device__ __forceinline__ float atomicMaxF(float* addr, float val) {
  int* ai = (int*)addr;
  int old = *ai;
  while (__int_as_float(old) < val) {
    int assumed = old;
    old = atomicCAS(ai, assumed, __float_as_int(val));
    if (old == assumed) break;
  }
  return __int_as_float(old);
}

__global__ void ge_edge_max(const float* __restrict__ als, const float* __restrict__ ald,
                            const int* __restrict__ ei, float* __restrict__ m, int heads) {
  int e = blockIdx.x * blockDim.x + threadIdx.x;
  if (e >= E_TOT) return;
  int s, d; ge_edge(ei, e, s, d);
  for (int h = 0; h < heads; ++h) {
    float v = ge_leaky(als[(size_t)s * heads + h] + ald[(size_t)d * heads + h]);
    atomicMaxF(&m[(size_t)d * heads + h], v);
  }
}

__global__ void ge_edge_expsum(const float* __restrict__ als, const float* __restrict__ ald,
                               const int* __restrict__ ei, const float* __restrict__ m,
                               float* __restrict__ ssum, float* __restrict__ alpha, int heads) {
  int e = blockIdx.x * blockDim.x + threadIdx.x;
  if (e >= E_TOT) return;
  int s, d; ge_edge(ei, e, s, d);
  for (int h = 0; h < heads; ++h) {
    float v = ge_leaky(als[(size_t)s * heads + h] + ald[(size_t)d * heads + h]);
    float p = expf(v - m[(size_t)d * heads + h]);
    alpha[(size_t)e * heads + h] = p;
    atomicAdd(&ssum[(size_t)d * heads + h], p);
  }
}

// thread per (edge, channel); loops heads
__global__ void ge_edge_scatter(const float* __restrict__ H, const float* __restrict__ alpha,
                                const float* __restrict__ ssum, const int* __restrict__ ei,
                                float* __restrict__ out, int heads) {
  long long i = (long long)blockIdx.x * blockDim.x + threadIdx.x;
  if (i >= (long long)E_TOT * HID) return;
  int c = (int)(i & (HID - 1));
  int e = (int)(i >> 6);
  int s, d; ge_edge(ei, e, s, d);
  for (int h = 0; h < heads; ++h) {
    float a = alpha[(size_t)e * heads + h] / (ssum[(size_t)d * heads + h] + 1e-16f);
    atomicAdd(&out[((size_t)d * heads + h) * HID + c],
              H[((size_t)s * heads + h) * HID + c] * a);
  }
}

// x = elu(in + b); write f16 always, f32 optionally
__global__ void ge_bias_elu(const float* __restrict__ in, const float* __restrict__ b,
                            _Float16* __restrict__ outh, float* __restrict__ outf,
                            long long n, int C) {
  long long i = (long long)blockIdx.x * blockDim.x + threadIdx.x;
  if (i >= n) return;
  float v = in[i] + b[(int)(i % C)];
  v = v > 0.f ? v : (expf(v) - 1.0f);
  outh[i] = (_Float16)v;
  if (outf) outf[i] = v;
}

__global__ void ge_gate_dot(const float* __restrict__ g1, const float* __restrict__ gw2,
                            const float* __restrict__ gb2, float* __restrict__ gate, int Nn) {
  int n = blockIdx.x * blockDim.x + threadIdx.x;
  if (n >= Nn) return;
  float s = gb2[0];
  const float* gp = g1 + (size_t)n * HID;
  for (int c = 0; c < HID; ++c) s += gp[c] * gw2[c];
  gate[n] = s;
}

__global__ void ge_gmax(const float* __restrict__ gate, float* gmax, int Nn) {
  int n = blockIdx.x * blockDim.x + threadIdx.x;
  if (n < Nn) atomicMaxF(gmax, gate[n]);
}
__global__ void ge_gsum(const float* __restrict__ gate, const float* gmax, float* gsum, int Nn) {
  int n = blockIdx.x * blockDim.x + threadIdx.x;
  if (n < Nn) atomicAdd(gsum, expf(gate[n] - *gmax));
}
__global__ void ge_wacc(const float* __restrict__ gate, const float* gmax,
                        const float* __restrict__ x3, float* acc, int Nn) {
  long long i = (long long)blockIdx.x * blockDim.x + threadIdx.x;
  if (i >= (long long)Nn * HID) return;
  int n = (int)(i >> 6), c = (int)(i & (HID - 1));
  atomicAdd(&acc[c], expf(gate[n] - *gmax) * x3[i]);
}
__global__ void ge_final(const float* acc, const float* gsum, float* out) {
  int c = threadIdx.x;
  out[c] = acc[c] / (*gsum);
}

static inline int blks(long long n, int t = 256) { return (int)((n + t - 1) / t); }

extern "C" void kernel_launch(void* const* d_in, const int* in_sizes, int n_in,
                              void* d_out, int out_size, void* d_ws, size_t ws_size,
                              hipStream_t stream) {
  const float* X    = (const float*)d_in[0];
  const int*   EI   = (const int*)  d_in[1];
  const float* W1   = (const float*)d_in[2];
  const float* as1  = (const float*)d_in[3];
  const float* ad1  = (const float*)d_in[4];
  const float* b1   = (const float*)d_in[5];
  const float* W2   = (const float*)d_in[6];
  const float* as2  = (const float*)d_in[7];
  const float* ad2  = (const float*)d_in[8];
  const float* b2   = (const float*)d_in[9];
  const float* gW1  = (const float*)d_in[10];
  const float* gb1  = (const float*)d_in[11];
  const float* gW2  = (const float*)d_in[12];
  const float* gb2  = (const float*)d_in[13];
  float* out = (float*)d_out;

  char* base = (char*)d_ws;
  size_t off = 0;
  auto alloc = [&](size_t bytes) -> void* {
    void* p = base + off;
    off += (bytes + 255) & ~(size_t)255;
    return p;
  };
  _Float16* Xh    = (_Float16*)alloc((size_t)N_NODES * EMB * 2);
  _Float16* W1t   = (_Float16*)alloc((size_t)C1 * EMB * 2);
  _Float16* W2t   = (_Float16*)alloc((size_t)HID * C1 * 2);
  _Float16* GW1t  = (_Float16*)alloc((size_t)HID * HID * 2);
  float*    H1    = (float*)   alloc((size_t)N_NODES * C1 * 4);
  float*    als1  = (float*)   alloc((size_t)N_NODES * HEADS * 4);
  float*    ald1  = (float*)   alloc((size_t)N_NODES * HEADS * 4);
  float*    m1    = (float*)   alloc((size_t)N_NODES * HEADS * 4);
  float*    s1    = (float*)   alloc((size_t)N_NODES * HEADS * 4);
  float*    alp1  = (float*)   alloc((size_t)E_TOT * HEADS * 4);
  float*    out1  = (float*)   alloc((size_t)N_NODES * C1 * 4);
  _Float16* x2h   = (_Float16*)alloc((size_t)N_NODES * C1 * 2);
  float*    H2    = (float*)   alloc((size_t)N_NODES * HID * 4);
  float*    als2  = (float*)   alloc((size_t)N_NODES * 4);
  float*    ald2  = (float*)   alloc((size_t)N_NODES * 4);
  float*    m2    = (float*)   alloc((size_t)N_NODES * 4);
  float*    s2    = (float*)   alloc((size_t)N_NODES * 4);
  float*    alp2  = (float*)   alloc((size_t)E_TOT * 4);
  float*    out2  = (float*)   alloc((size_t)N_NODES * HID * 4);
  _Float16* x3h   = (_Float16*)alloc((size_t)N_NODES * HID * 2);
  float*    x3f   = (float*)   alloc((size_t)N_NODES * HID * 4);
  float*    G1    = (float*)   alloc((size_t)N_NODES * HID * 4);
  float*    gate  = (float*)   alloc((size_t)N_NODES * 4);
  float*    gmax  = (float*)   alloc(4);
  float*    gsum  = (float*)   alloc(4);
  float*    acc   = (float*)   alloc(64 * 4);

  const float NEG = -3.402823466e+38f;
  const int MT = N_NODES / 16;            // 3125 row tiles
  const int GX = (MT + 7) / 8;            // 8 waves per block

  // ---- precision conversion / weight packing ----
  ge_cvt_f16<<<blks((long long)N_NODES * EMB), 256, 0, stream>>>(X, Xh, (long long)N_NODES * EMB);
  ge_pack_wt<<<blks((long long)EMB * C1), 256, 0, stream>>>(W1, W1t, EMB, C1);
  ge_pack_wt<<<blks((long long)C1 * HID), 256, 0, stream>>>(W2, W2t, C1, HID);
  ge_pack_wt<<<blks((long long)HID * HID), 256, 0, stream>>>(gW1, GW1t, HID, HID);

  // ---- GAT layer 1 ----
  ge_gemm_wmma<<<dim3(GX, C1 / 64), 256, 0, stream>>>(Xh, W1t, H1, nullptr,
                                                      N_NODES, C1, EMB, 0);
  ge_logits<<<blks((long long)N_NODES * HEADS), 256, 0, stream>>>(H1, as1, ad1, als1, ald1,
                                                                  N_NODES, HEADS);
  ge_fill<<<blks((long long)N_NODES * HEADS), 256, 0, stream>>>(m1, NEG, (long long)N_NODES * HEADS);
  ge_fill<<<blks((long long)N_NODES * HEADS), 256, 0, stream>>>(s1, 0.f, (long long)N_NODES * HEADS);
  ge_fill<<<blks((long long)N_NODES * C1), 256, 0, stream>>>(out1, 0.f, (long long)N_NODES * C1);
  ge_edge_max<<<blks(E_TOT), 256, 0, stream>>>(als1, ald1, EI, m1, HEADS);
  ge_edge_expsum<<<blks(E_TOT), 256, 0, stream>>>(als1, ald1, EI, m1, s1, alp1, HEADS);
  ge_edge_scatter<<<blks((long long)E_TOT * HID), 256, 0, stream>>>(H1, alp1, s1, EI, out1, HEADS);
  ge_bias_elu<<<blks((long long)N_NODES * C1), 256, 0, stream>>>(out1, b1, x2h, nullptr,
                                                                 (long long)N_NODES * C1, C1);

  // ---- GAT layer 2 (1 head) ----
  ge_gemm_wmma<<<dim3(GX, 1), 256, 0, stream>>>(x2h, W2t, H2, nullptr,
                                                N_NODES, HID, C1, 0);
  ge_logits<<<blks(N_NODES), 256, 0, stream>>>(H2, as2, ad2, als2, ald2, N_NODES, 1);
  ge_fill<<<blks(N_NODES), 256, 0, stream>>>(m2, NEG, N_NODES);
  ge_fill<<<blks(N_NODES), 256, 0, stream>>>(s2, 0.f, N_NODES);
  ge_fill<<<blks((long long)N_NODES * HID), 256, 0, stream>>>(out2, 0.f, (long long)N_NODES * HID);
  ge_edge_max<<<blks(E_TOT), 256, 0, stream>>>(als2, ald2, EI, m2, 1);
  ge_edge_expsum<<<blks(E_TOT), 256, 0, stream>>>(als2, ald2, EI, m2, s2, alp2, 1);
  ge_edge_scatter<<<blks((long long)E_TOT * HID), 256, 0, stream>>>(H2, alp2, s2, EI, out2, 1);
  ge_bias_elu<<<blks((long long)N_NODES * HID), 256, 0, stream>>>(out2, b2, x3h, x3f,
                                                                  (long long)N_NODES * HID, HID);

  // ---- attentional pooling ----
  ge_gemm_wmma<<<dim3(GX, 1), 256, 0, stream>>>(x3h, GW1t, G1, gb1,
                                                N_NODES, HID, HID, 1);
  ge_gate_dot<<<blks(N_NODES), 256, 0, stream>>>(G1, gW2, gb2, gate, N_NODES);
  ge_fill<<<1, 1, 0, stream>>>(gmax, NEG, 1);
  ge_fill<<<1, 1, 0, stream>>>(gsum, 0.f, 1);
  ge_fill<<<1, 64, 0, stream>>>(acc, 0.f, 64);
  ge_gmax<<<blks(N_NODES), 256, 0, stream>>>(gate, gmax, N_NODES);
  ge_gsum<<<blks(N_NODES), 256, 0, stream>>>(gate, gmax, gsum, N_NODES);
  ge_wacc<<<blks((long long)N_NODES * HID), 256, 0, stream>>>(gate, gmax, x3f, acc, N_NODES);
  ge_final<<<1, 64, 0, stream>>>(acc, gsum, out);
}
